// CMTA_58884001628669
// MI455X (gfx1250) — matrix operand order repaired and checked by reading, not compiled, not run on hardware
//
#include <hip/hip_runtime.h>
#include <hip/hip_bf16.h>
#include <stdint.h>

typedef __attribute__((ext_vector_type(16))) __bf16 v16bf;
typedef __attribute__((ext_vector_type(8)))  float  v8f;

#define D_DIM 512
#define E_DIM 8
#define MT    128         // tokens per block
#define BLK   512         // threads per block (16 waves)
#define NWAVE 16
#define XS_STRIDE  520    // bf16 row stride (pad to dodge bank conflicts)

union FragU { uint4 u[2]; v16bf v; };

// A fragment (16x32 bf16) from LDS, row-major buffer.
// ISA layout: lane%16 = row M; lanes 0-15: V0-3=K0..7, V4-7=K16..23;
//             lanes 16-31: V0-3=K8..15, V4-7=K24..31.
__device__ __forceinline__ v16bf load_frag_a(const __bf16* buf, int row, int k0,
                                             int lane, int rs) {
  const int base = (lane < 16) ? 0 : 8;
  FragU f;
  f.u[0] = *(const uint4*)(buf + row * rs + k0 + base);
  f.u[1] = *(const uint4*)(buf + row * rs + k0 + 16 + base);
  return f.v;
}

// B fragment (32x16 bf16) from pre-swizzled global weights:
// each fragment = 512 bf16, lane's 16 values contiguous.
__device__ __forceinline__ v16bf load_frag_b(const __bf16* wsw, int frag, int lane) {
  const __bf16* p = wsw + ((size_t)frag << 9) + (lane << 4);
  FragU f;
  f.u[0] = ((const uint4*)p)[0];
  f.u[1] = ((const uint4*)p)[1];
  return f.v;
}

__global__ void CMTA_zero_kernel(float* p, int n) {
  int i = blockIdx.x * blockDim.x + threadIdx.x;
  if (i < n) p[i] = 0.f;
}

// Convert W (E,D,D) f32 row-major (row = K, col = N) into per-fragment
// swizzled bf16. frag index = ((e*16 + kc)*32 + nt); within fragment,
// lane l holds 16 contiguous bf16: K = kc*32 + (l<16 ? 0 : 8) + (j<8 ? j : 8+j), N = nt*16 + l%16.
__global__ void CMTA_swz_kernel(const float* __restrict__ W, __bf16* __restrict__ Wsw) {
  int idx = blockIdx.x * blockDim.x + threadIdx.x;   // one thread per (e,kc,nt,lane)
  if (idx >= E_DIM * 16 * 32 * 32) return;
  int l  = idx & 31;
  int nt = (idx >> 5) & 31;
  int kc = (idx >> 10) & 15;
  int e  = idx >> 14;
  int n     = nt * 16 + (l & 15);
  int kbase = kc * 32 + ((l < 16) ? 0 : 8);
  __bf16* dst = Wsw + ((size_t)idx << 4);
  const float* src = W + (size_t)e * D_DIM * D_DIM + (size_t)kbase * D_DIM + n;
#pragma unroll
  for (int j = 0; j < 16; ++j) dst[j] = (__bf16)src[(size_t)(j + (j >= 8 ? 8 : 0)) * D_DIM];
}

// Gating: one wave per token -> dense top-2 / bottom-2 softmax combine weights.
__global__ __launch_bounds__(256) void CMTA_gate_kernel(
    const float* __restrict__ x, const float* __restrict__ Wg,
    const float* __restrict__ bg, float* __restrict__ wtop,
    float* __restrict__ wbot, int T) {
  int wave = threadIdx.x >> 5;
  int lane = threadIdx.x & 31;
  int t = blockIdx.x * 8 + wave;
  if (t >= T) return;
  float p[E_DIM] = {};
  const float* xr = x + (size_t)t * D_DIM;
  for (int d = lane; d < D_DIM; d += 32) {
    float xv = xr[d];
#pragma unroll
    for (int e = 0; e < E_DIM; ++e) p[e] += xv * Wg[d * E_DIM + e];
  }
#pragma unroll
  for (int e = 0; e < E_DIM; ++e)
    for (int off = 16; off; off >>= 1) p[e] += __shfl_down(p[e], off);
  if (lane == 0) {
    float s[E_DIM];
#pragma unroll
    for (int e = 0; e < E_DIM; ++e) s[e] = p[e] + bg[e];
    int i0 = 0; for (int e = 1; e < E_DIM; ++e) if (s[e] > s[i0]) i0 = e;
    int i1 = -1;
    for (int e = 0; e < E_DIM; ++e) { if (e == i0) continue; if (i1 < 0 || s[e] > s[i1]) i1 = e; }
    int j0 = 0; for (int e = 1; e < E_DIM; ++e) if (s[e] < s[j0]) j0 = e;
    int j1 = -1;
    for (int e = 0; e < E_DIM; ++e) { if (e == j0) continue; if (j1 < 0 || s[e] < s[j1]) j1 = e; }
    // softmax over the 2 selected (top: s[i0] >= s[i1]; bottom: s[j0] <= s[j1])
    float et1 = __expf(s[i1] - s[i0]);
    float invt = 1.f / (1.f + et1);
    float eb0 = __expf(s[j0] - s[j1]);
    float invb = 1.f / (eb0 + 1.f);
    float wt[E_DIM] = {}, wb[E_DIM] = {};
    wt[i0] = invt;        wt[i1] = et1 * invt;
    wb[j0] = eb0 * invb;  wb[j1] = invb;
#pragma unroll
    for (int e = 0; e < E_DIM; ++e) {
      wtop[(size_t)t * E_DIM + e] = wt[e];
      wbot[(size_t)t * E_DIM + e] = wb[e];
    }
  }
}

__global__ __launch_bounds__(BLK, 1) void CMTA_moe_kernel(
    const float* __restrict__ x,
    const __bf16* __restrict__ w1sw, const __bf16* __restrict__ w2sw,
    const float* __restrict__ b1, const float* __restrict__ ln1w,
    const float* __restrict__ ln1b, const float* __restrict__ b2,
    const float* __restrict__ ln2w, const float* __restrict__ ln2b,
    const float* __restrict__ wtop, const float* __restrict__ wbot,
    float* __restrict__ out, float* __restrict__ nacc, int T) {
  __shared__ __bf16 xs[MT * XS_STRIDE];      // 130 KB  x tile (bf16)
  __shared__ __bf16 hstage[MT * XS_STRIDE];  // 130 KB  activation staging (bf16)
  __shared__ float  racc[NWAVE];

  const int tid  = threadIdx.x;
  const int lane = tid & 31;
  const int wave = tid >> 5;
  const int wm   = wave & 3;   // token group: rows [wm*32, wm*32+32) = 2 M-tiles
  const int wn   = wave >> 2;  // 128-column group = 8 N-tiles
  const int t0   = blockIdx.x * MT;

  float* outTop = out + (size_t)T * D_DIM;
  float* outBot = out + (size_t)2 * T * D_DIM;

  // stage x -> bf16 LDS
  for (int i = tid; i < MT * D_DIM; i += BLK) {
    int r = i >> 9, c = i & 511;
    xs[r * XS_STRIDE + c] = (__bf16)x[(size_t)(t0 + r) * D_DIM + c];
  }
  __syncthreads();

  const v8f vzero = {0.f, 0.f, 0.f, 0.f, 0.f, 0.f, 0.f, 0.f};

  for (int e = 0; e < E_DIM; ++e) {
    // ---------------- layer 1: h = xs @ W1[e] ----------------
    v8f acc[2][8];
#pragma unroll
    for (int mi = 0; mi < 2; ++mi)
#pragma unroll
      for (int i = 0; i < 8; ++i) acc[mi][i] = vzero;
    for (int kc = 0; kc < 16; ++kc) {
      v16bf a0 = load_frag_a(xs, wm * 32 + (lane & 15), kc * 32, lane, XS_STRIDE);
      v16bf a1 = load_frag_a(xs, wm * 32 + 16 + (lane & 15), kc * 32, lane, XS_STRIDE);
      int fragBase = (e * 16 + kc) * 32 + wn * 8;
      __builtin_prefetch(w1sw + ((size_t)(fragBase + 32) << 9), 0, 1);
#pragma unroll
      for (int nt = 0; nt < 8; ++nt) {
        v16bf b = load_frag_b(w1sw, fragBase + nt, lane);
        acc[0][nt] = __builtin_amdgcn_wmma_f32_16x16x32_bf16(
            false, a0, false, b, (short)0, acc[0][nt], false, false);
        acc[1][nt] = __builtin_amdgcn_wmma_f32_16x16x32_bf16(
            false, a1, false, b, (short)0, acc[1][nt], false, false);
      }
    }
    // C layout: elem r of lane l -> row = r + (l<16?0:8), col = lane%16
#pragma unroll
    for (int nt = 0; nt < 8; ++nt) {
      int col = wn * 128 + nt * 16 + (lane & 15);
      float bias = b1[e * D_DIM + col];
#pragma unroll
      for (int mi = 0; mi < 2; ++mi)
#pragma unroll
        for (int r = 0; r < 8; ++r) {
          int row = wm * 32 + mi * 16 + r + ((lane < 16) ? 0 : 8);
          hstage[row * XS_STRIDE + col] = (__bf16)(acc[mi][nt][r] + bias);
        }
    }
    __syncthreads();

    // ---------------- LN1 + ReLU, in place on hstage (rows disjoint per group) ----
    {
      int t = tid >> 2, g = tid & 3;   // 4 threads per token
      float s = 0.f, s2 = 0.f;
      for (int i = g; i < D_DIM; i += 4) {
        float v = (float)hstage[t * XS_STRIDE + i];
        s += v; s2 += v * v;
      }
      s += __shfl_xor(s, 1); s2 += __shfl_xor(s2, 1);
      s += __shfl_xor(s, 2); s2 += __shfl_xor(s2, 2);
      float mu = s * (1.f / D_DIM);
      float var = s2 * (1.f / D_DIM) - mu * mu;
      float rstd = rsqrtf(var + 1e-5f);
      for (int i = g; i < D_DIM; i += 4) {
        float v = ((float)hstage[t * XS_STRIDE + i] - mu) * rstd * ln1w[e * D_DIM + i]
                  + ln1b[e * D_DIM + i];
        hstage[t * XS_STRIDE + i] = (__bf16)fmaxf(v, 0.f);
      }
    }
    __syncthreads();

    // ---------------- layer 2: h2 = hstage @ W2[e] ----------------
#pragma unroll
    for (int mi = 0; mi < 2; ++mi)
#pragma unroll
      for (int i = 0; i < 8; ++i) acc[mi][i] = vzero;
    for (int kc = 0; kc < 16; ++kc) {
      v16bf a0 = load_frag_a(hstage, wm * 32 + (lane & 15), kc * 32, lane, XS_STRIDE);
      v16bf a1 = load_frag_a(hstage, wm * 32 + 16 + (lane & 15), kc * 32, lane, XS_STRIDE);
      int fragBase = (e * 16 + kc) * 32 + wn * 8;
      __builtin_prefetch(w2sw + ((size_t)(fragBase + 32) << 9), 0, 1);
#pragma unroll
      for (int nt = 0; nt < 8; ++nt) {
        v16bf b = load_frag_b(w2sw, fragBase + nt, lane);
        acc[0][nt] = __builtin_amdgcn_wmma_f32_16x16x32_bf16(
            false, a0, false, b, (short)0, acc[0][nt], false, false);
        acc[1][nt] = __builtin_amdgcn_wmma_f32_16x16x32_bf16(
            false, a1, false, b, (short)0, acc[1][nt], false, false);
      }
    }
    __syncthreads();   // all waves done reading hstage before overwriting it
#pragma unroll
    for (int nt = 0; nt < 8; ++nt) {
      int col = wn * 128 + nt * 16 + (lane & 15);
      float bias = b2[e * D_DIM + col];
#pragma unroll
      for (int mi = 0; mi < 2; ++mi)
#pragma unroll
        for (int r = 0; r < 8; ++r) {
          int row = wm * 32 + mi * 16 + r + ((lane < 16) ? 0 : 8);
          hstage[row * XS_STRIDE + col] = (__bf16)(acc[mi][nt][r] + bias);
        }
    }
    __syncthreads();

    // ---------------- LN2 + weighted combine into out_top/out_bot ----------------
    {
      int t = tid >> 2, g = tid & 3;
      float s = 0.f, s2 = 0.f;
      for (int i = g; i < D_DIM; i += 4) {
        float v = (float)hstage[t * XS_STRIDE + i];
        s += v; s2 += v * v;
      }
      s += __shfl_xor(s, 1); s2 += __shfl_xor(s2, 1);
      s += __shfl_xor(s, 2); s2 += __shfl_xor(s2, 2);
      float mu = s * (1.f / D_DIM);
      float var = s2 * (1.f / D_DIM) - mu * mu;
      float rstd = rsqrtf(var + 1e-5f);
      float gt = wtop[(size_t)(t0 + t) * E_DIM + e];
      float gb = wbot[(size_t)(t0 + t) * E_DIM + e];
      size_t rowoff = (size_t)(t0 + t) * D_DIM;
      for (int i = g; i < D_DIM; i += 4) {
        float v = ((float)hstage[t * XS_STRIDE + i] - mu) * rstd * ln2w[e * D_DIM + i]
                  + ln2b[e * D_DIM + i];
        if (e == 0) {
          outTop[rowoff + i] = gt * v;
          outBot[rowoff + i] = gb * v;
        } else {
          outTop[rowoff + i] += gt * v;
          outBot[rowoff + i] += gb * v;
        }
      }
    }
    __syncthreads();
  }

  // ---------------- output = out_top + x ; orth-loss partials ----------------
  float dacc = 0.f;
  for (int i = tid; i < MT * D_DIM; i += BLK) {
    int r = i >> 9, c = i & 511;
    size_t idx = (size_t)(t0 + r) * D_DIM + c;
    float tv = outTop[idx], bv = outBot[idx];
    out[idx] = tv + x[idx];
    float d = tv - bv;
    dacc += d * d;
  }
  for (int off = 16; off; off >>= 1) dacc += __shfl_down(dacc, off);
  if (lane == 0) racc[wave] = dacc;
  __syncthreads();
  if (tid == 0) {
    float s = 0.f;
#pragma unroll
    for (int i = 0; i < NWAVE; ++i) s += racc[i];
    atomicAdd(&nacc[t0 / 4096], s);   // N = 4096 tokens per batch
  }
}

__global__ void CMTA_loss_kernel(const float* __restrict__ nacc, float* __restrict__ outl) {
  if (blockIdx.x == 0 && threadIdx.x == 0) {
    float s = 0.f;
#pragma unroll
    for (int b = 0; b < E_DIM; ++b) s += 1.f / (sqrtf(nacc[b]) + 1e-8f);
    *outl = s * (1.f / 8.f);
  }
}

extern "C" void kernel_launch(void* const* d_in, const int* in_sizes, int n_in,
                              void* d_out, int out_size, void* d_ws, size_t ws_size,
                              hipStream_t stream) {
  const float* x    = (const float*)d_in[0];
  const float* Wg   = (const float*)d_in[1];
  const float* bg   = (const float*)d_in[2];
  const float* W1   = (const float*)d_in[3];
  const float* b1   = (const float*)d_in[4];
  const float* ln1w = (const float*)d_in[5];
  const float* ln1b = (const float*)d_in[6];
  const float* W2   = (const float*)d_in[7];
  const float* b2   = (const float*)d_in[8];
  const float* ln2w = (const float*)d_in[9];
  const float* ln2b = (const float*)d_in[10];
  // d_in[11] = k (==2), baked in.

  const int T = in_sizes[0] / D_DIM;   // B*N = 32768 tokens

  char* ws = (char*)d_ws;
  __bf16* w1sw = (__bf16*)ws;                                 // 4 MB
  __bf16* w2sw = (__bf16*)(ws + (size_t)(4u << 20));          // 4 MB
  float*  wtop = (float*)(ws + (size_t)(8u << 20));           // 1 MB
  float*  wbot = (float*)(ws + (size_t)(8u << 20) + (size_t)T * E_DIM * 4);
  float*  nacc = (float*)(ws + (size_t)(8u << 20) + (size_t)2 * T * E_DIM * 4);

  float* out = (float*)d_out;

  CMTA_zero_kernel<<<1, 32, 0, stream>>>(nacc, E_DIM);
  const int swzThreads = E_DIM * 16 * 32 * 32;  // 131072
  CMTA_swz_kernel<<<swzThreads / 256, 256, 0, stream>>>(W1, w1sw);
  CMTA_swz_kernel<<<swzThreads / 256, 256, 0, stream>>>(W2, w2sw);
  CMTA_gate_kernel<<<T / 8, 256, 0, stream>>>(x, Wg, bg, wtop, wbot, T);
  CMTA_moe_kernel<<<T / MT, BLK, 0, stream>>>(x, w1sw, w2sw, b1, ln1w, ln1b,
                                              b2, ln2w, ln2b, wtop, wbot,
                                              out, nacc, T);
  CMTA_loss_kernel<<<1, 32, 0, stream>>>(nacc, out + (size_t)3 * T * D_DIM);
}
